// AttnHead_41841571398084
// MI455X (gfx1250) — matrix operand order repaired and checked by reading, not compiled, not run on hardware
//
#include <hip/hip_runtime.h>
#include <hip/hip_bf16.h>
#include <cmath>

// Problem constants (match reference).
#define B_ 8
#define T_ 2048
#define C_ 2048
#define H_ 2048

typedef __bf16 bhalf;
typedef __bf16 bf16x16 __attribute__((ext_vector_type(16)));
typedef __bf16 bf16x8  __attribute__((ext_vector_type(8)));
typedef __bf16 bf16x4  __attribute__((ext_vector_type(4)));
typedef float  f32x8   __attribute__((ext_vector_type(8)));
typedef float  f32x4   __attribute__((ext_vector_type(4)));

// float -> bf16 round-to-nearest-even via bit manipulation (no fp-env deps).
__device__ __forceinline__ bhalf f2bf(float f) {
  unsigned u = __builtin_bit_cast(unsigned, f);
  unsigned r = (u + 0x7FFFu + ((u >> 16) & 1u)) >> 16;
  return __builtin_bit_cast(bhalf, (unsigned short)r);
}

// ---- WMMA fragment loaders (CDNA5 wave32 layouts, CDNA5 ISA) ----
// A: 16x32 (MxK) bf16, source row-major [M x lda].
// lanes 0-15: M=lane, K = 0..7 / 16..23 ; lanes 16-31: M=lane-16, K = 8..15 / 24..31
__device__ __forceinline__ bf16x16 load_a(const bhalf* base, int lda) {
  const int lane = threadIdx.x & 31;
  const int row  = lane & 15;
  const int k0   = (lane >> 4) << 3;           // 0 or 8
  const bhalf* p = base + (size_t)row * lda + k0;
  bf16x8 lo = *(const bf16x8*)p;               // K = k0 .. k0+7
  bf16x8 hi = *(const bf16x8*)(p + 16);        // K = k0+16 .. k0+23
  bf16x16 r;
#pragma unroll
  for (int i = 0; i < 8; ++i) { r[i] = lo[i]; r[i + 8] = hi[i]; }
  return r;
}

// B: 32x16 (KxN) bf16, sourced from an N-major buffer [N x ldb] (K contiguous).
// lanes 0-15: N=lane, K = 0..7 / 16..23 ; lanes 16-31: N=lane-16, K = 8..15 / 24..31.
__device__ __forceinline__ bf16x16 load_b(const bhalf* baseNK, int ldb) {
  const int lane = threadIdx.x & 31;
  const int col  = lane & 15;
  const int k0   = (lane >> 4) << 3;           // 0 or 8
  const bhalf* p = baseNK + (size_t)col * ldb + k0;
  bf16x8 lo = *(const bf16x8*)p;               // K = k0 .. k0+7
  bf16x8 hi = *(const bf16x8*)(p + 16);        // K = k0+16 .. k0+23
  bf16x16 r;
#pragma unroll
  for (int i = 0; i < 8; ++i) { r[i] = lo[i]; r[i + 8] = hi[i]; }
  return r;
}

__device__ __forceinline__ f32x8 wmma_bf16(bf16x16 a, bf16x16 b, f32x8 c) {
  return __builtin_amdgcn_wmma_f32_16x16x32_bf16(false, a, false, b, (short)0, c,
                                                 false, false);
}

__device__ __forceinline__ void zero_acc(f32x8 acc[2][4]) {
#pragma unroll
  for (int i = 0; i < 2; ++i)
#pragma unroll
    for (int j = 0; j < 4; ++j)
#pragma unroll
      for (int e = 0; e < 8; ++e) acc[i][j][e] = 0.0f;
}

// ------------------- CDNA5 async-to-LDS staged block GEMM -------------------
// Block tile 128(M) x 128(N), K staged 64 at a time through LDS, double
// buffered with GLOBAL_LOAD_ASYNC_TO_LDS_B128 (ASYNCcnt) + workgroup barriers.
// LDS row stride padded to 72 bf16 (36 words): lane l reads at word 36*l mod 64
// -> the 16 b128 spans are pairwise disjoint banks (conflict-free ds_load_b128).
#define LDS_K 64
#define LDS_STRIDE 72

__device__ __forceinline__ void async_b128(unsigned lds_byte, const bhalf* g) {
  asm volatile("global_load_async_to_lds_b128 %0, %1, off"
               :: "v"(lds_byte), "v"(g)
               : "memory");
}

// Copy one 128x64 bf16 tile of A and one of B into LDS (async DMA).
// 256 threads * 4 chunks * 16B = 16KB per tile; 8 async instrs per wave.
__device__ __forceinline__ void issue_stage(const bhalf* __restrict__ Ag,
                                            const bhalf* __restrict__ Bg,
                                            int ld, int k0,
                                            bhalf* Al, bhalf* Bl) {
  const unsigned abase = (unsigned)(size_t)Al;   // low 32 bits = LDS offset
  const unsigned bbase = (unsigned)(size_t)Bl;
#pragma unroll
  for (int i = 0; i < 4; ++i) {
    const int c    = (int)threadIdx.x + (i << 8);
    const int row  = c >> 3;                     // 8 x 16B chunks per 64-el row
    const int colE = (c & 7) << 3;
    const unsigned loff = (unsigned)(row * LDS_STRIDE + colE) * 2u;
    async_b128(abase + loff, Ag + (size_t)row * ld + k0 + colE);
    async_b128(bbase + loff, Bg + (size_t)row * ld + k0 + colE);
  }
}

__device__ __forceinline__ void compute_stage(const bhalf* Al, const bhalf* Bl,
                                              int wm, int wn, f32x8 acc[2][4]) {
#pragma unroll
  for (int kk = 0; kk < LDS_K; kk += 32) {
    bf16x16 a0 = load_a(Al + (size_t)(wm * 32) * LDS_STRIDE + kk, LDS_STRIDE);
    bf16x16 a1 = load_a(Al + (size_t)(wm * 32 + 16) * LDS_STRIDE + kk, LDS_STRIDE);
#pragma unroll
    for (int j = 0; j < 4; ++j) {
      bf16x16 b =
          load_b(Bl + (size_t)(wn * 64 + 16 * j) * LDS_STRIDE + kk, LDS_STRIDE);
      acc[0][j] = wmma_bf16(a0, b, acc[0][j]);
      acc[1][j] = wmma_bf16(a1, b, acc[1][j]);
    }
  }
}

// Ag/Bg are the BLOCK origins (row m0b of A, row n0b of N-major B), ld = row
// stride, kLen multiple of 64 and uniform across the block.
__device__ __forceinline__ void gemm_block_lds(const bhalf* __restrict__ Ag,
                                               const bhalf* __restrict__ Bg,
                                               int ld, int kLen,
                                               int wm, int wn, f32x8 acc[2][4]) {
  __shared__ bhalf Al[2][128 * LDS_STRIDE];
  __shared__ bhalf Bl[2][128 * LDS_STRIDE];
  const int ntiles = kLen >> 6;
  issue_stage(Ag, Bg, ld, 0, Al[0], Bl[0]);
  for (int kt = 0; kt < ntiles; ++kt) {
    const int cur = kt & 1;
    if (kt + 1 < ntiles) {
      issue_stage(Ag, Bg, ld, (kt + 1) << 6, Al[cur ^ 1], Bl[cur ^ 1]);
      // 8 newer async loads may remain in flight; stage kt is complete.
      asm volatile("s_wait_asynccnt 0x8" ::: "memory");
    } else {
      asm volatile("s_wait_asynccnt 0x0" ::: "memory");
    }
    __syncthreads();                 // stage kt visible to all waves
    compute_stage(Al[cur], Bl[cur], wm, wn, acc);
    __syncthreads();                 // all reads done before buffer reuse
  }
}

// ---------------- Kernel 0a: x fp32 -> bf16 (xb fits in 192MB L2) ------------
__global__ void k_cvt_bf16(const float* __restrict__ in, bhalf* __restrict__ out,
                           int n) {
  int i = (blockIdx.x * blockDim.x + threadIdx.x) * 4;
  if (i + 3 < n) {
    f32x4 v = *(const f32x4*)(in + i);
    bf16x4 o;
#pragma unroll
    for (int j = 0; j < 4; ++j) o[j] = f2bf(v[j]);
    *(bf16x4*)(out + i) = o;
  }
}

// ------------- Kernel 0b: W [C,H] fp32 -> W^T [H,C] bf16 (LDS tiled) ---------
__global__ void k_transpose_w(const float* __restrict__ w, bhalf* __restrict__ wt) {
  __shared__ float tile[32][33];
  const int c0 = blockIdx.y * 32, h0 = blockIdx.x * 32;
  const int tx = threadIdx.x, ty = threadIdx.y;   // block (32, 8)
#pragma unroll
  for (int i = 0; i < 32; i += 8)
    tile[ty + i][tx] = w[(size_t)(c0 + ty + i) * H_ + h0 + tx];
  __syncthreads();
#pragma unroll
  for (int i = 0; i < 32; i += 8)
    wt[(size_t)(h0 + ty + i) * C_ + c0 + tx] = f2bf(tile[tx][ty + i]);
}

// ------------- Kernel 1: projection GEMM  out = xb @ wT' + bias --------------
// xb: [M=B*T, C] bf16.  wT: [H, C] bf16 (N-major).  q/k: row-major [M,H] bf16.
// v: stored transposed [B, H, T] bf16 (natural B-operand of O = P*V; stores
// become contiguous 16B vectors).
__global__ __launch_bounds__(256)
void k_proj(const bhalf* __restrict__ xb, const bhalf* __restrict__ wT,
            const float* __restrict__ bias, bhalf* __restrict__ outRM,
            bhalf* __restrict__ outTB, int transposed) {
  const int wid  = threadIdx.x >> 5;
  const int wm   = wid & 3, wn = wid >> 2;        // 4 x 2 waves
  const int m0b  = blockIdx.y * 128;
  const int n0b  = blockIdx.x * 128;
  const int m0   = m0b + wm * 32;
  const int n0   = n0b + wn * 64;
  const int lane = threadIdx.x & 31;

  f32x8 acc[2][4];
  zero_acc(acc);
  gemm_block_lds(xb + (size_t)m0b * C_, wT + (size_t)n0b * C_, C_, C_, wm, wn,
                 acc);

  const int nl = lane & 15, mb = (lane >> 4) * 8;
#pragma unroll
  for (int i = 0; i < 2; ++i)
#pragma unroll
    for (int j = 0; j < 4; ++j) {
      const int n    = n0 + 16 * j + nl;
      const int mrow = m0 + 16 * i + mb;
      const float bv = bias[n];
      if (!transposed) {
#pragma unroll
        for (int r = 0; r < 8; ++r)
          outRM[(size_t)(mrow + r) * H_ + n] = f2bf(acc[i][j][r] + bv);
      } else {
        const int bidx = mrow / T_, t = mrow % T_;  // whole tile in one batch
        bf16x8 pk;
#pragma unroll
        for (int r = 0; r < 8; ++r) pk[r] = f2bf(acc[i][j][r] + bv);
        *(bf16x8*)(outTB + (size_t)bidx * H_ * T_ + (size_t)n * T_ + t) = pk;
      }
    }
}

// ------------- Kernel 2: S = scale * q @ k^T (lower-trapezoid blocks only) ---
__global__ __launch_bounds__(256)
void k_scores(const bhalf* __restrict__ qb, const bhalf* __restrict__ kb,
              float* __restrict__ S, float scale) {
  const int b   = blockIdx.z;
  const int m0b = blockIdx.y * 128;
  const int n0b = blockIdx.x * 128;
  if (n0b > m0b + 127) return;   // fully above causal diagonal: never read
  const bhalf* q = qb + (size_t)b * T_ * H_;
  const bhalf* k = kb + (size_t)b * T_ * H_;
  float* s = S + (size_t)b * T_ * T_;

  const int wid  = threadIdx.x >> 5;
  const int wm   = wid & 3, wn = wid >> 2;
  const int m0   = m0b + wm * 32;
  const int n0   = n0b + wn * 64;
  const int lane = threadIdx.x & 31;

  f32x8 acc[2][4];
  zero_acc(acc);
  // A = q rows (row-major [T,H]); B = k rows: [T,H] is exactly N-major (N=key).
  gemm_block_lds(q + (size_t)m0b * H_, k + (size_t)n0b * H_, H_, H_, wm, wn,
                 acc);

  const int nl = lane & 15, mb = (lane >> 4) * 8;
#pragma unroll
  for (int i = 0; i < 2; ++i)
#pragma unroll
    for (int j = 0; j < 4; ++j) {
      const int n = n0 + 16 * j + nl, mrow = m0 + 16 * i + mb;
#pragma unroll
      for (int r = 0; r < 8; ++r)
        s[(size_t)(mrow + r) * T_ + n] = acc[i][j][r] * scale;
    }
}

// ------------- Kernel 3: causal row softmax; P = softmax(S) as bf16 ----------
__global__ __launch_bounds__(256)
void k_softmax(const float* __restrict__ S, bhalf* __restrict__ P) {
  const int b = blockIdx.y;
  const int r = blockIdx.x;  // query row: valid keys are cols 0..r
  const float* srow = S + ((size_t)b * T_ + r) * T_;
  bhalf* prow = P + ((size_t)b * T_ + r) * T_;

  float vals[8];
  float mx = -INFINITY;
#pragma unroll
  for (int i = 0; i < 8; ++i) {
    const int c = threadIdx.x + i * 256;
    const float v = (c <= r) ? srow[c] : -INFINITY;
    vals[i] = v;
    mx = fmaxf(mx, v);
  }
#pragma unroll
  for (int off = 16; off; off >>= 1) mx = fmaxf(mx, __shfl_xor(mx, off, 32));
  __shared__ float redm[8], reds[8];
  const int wid = threadIdx.x >> 5;
  if ((threadIdx.x & 31) == 0) redm[wid] = mx;
  __syncthreads();
  float bm = redm[0];
#pragma unroll
  for (int i = 1; i < 8; ++i) bm = fmaxf(bm, redm[i]);

  float sum = 0.0f;
#pragma unroll
  for (int i = 0; i < 8; ++i) {
    const float e = __expf(vals[i] - bm);  // masked (-inf) -> 0
    vals[i] = e;
    sum += e;
  }
#pragma unroll
  for (int off = 16; off; off >>= 1) sum += __shfl_xor(sum, off, 32);
  if ((threadIdx.x & 31) == 0) reds[wid] = sum;
  __syncthreads();
  float bs = 0.0f;
#pragma unroll
  for (int i = 0; i < 8; ++i) bs += reds[i];
  const float inv = 1.0f / bs;
#pragma unroll
  for (int i = 0; i < 8; ++i) {
    const int c = threadIdx.x + i * 256;
    prow[c] = f2bf(vals[i] * inv);         // masked entries store exact 0
  }
}

// ------------- Kernel 4: O = P @ V, with causal K-loop cap -------------------
__global__ __launch_bounds__(256)
void k_out(const bhalf* __restrict__ P, const bhalf* __restrict__ vt,
           float* __restrict__ O) {
  const int b   = blockIdx.z;
  const int m0b = blockIdx.y * 128;
  const int n0b = blockIdx.x * 128;
  const bhalf* p = P + (size_t)b * T_ * T_;
  const bhalf* v = vt + (size_t)b * H_ * T_;    // [H, T]: N-major for B-operand
  float* o = O + (size_t)b * T_ * H_;

  const int wid  = threadIdx.x >> 5;
  const int wm   = wid & 3, wn = wid >> 2;
  const int m0   = m0b + wm * 32;
  const int n0   = n0b + wn * 64;
  const int lane = threadIdx.x & 31;
  const int kmax = m0b + 128;                   // keys > query are exactly 0

  f32x8 acc[2][4];
  zero_acc(acc);
  gemm_block_lds(p + (size_t)m0b * T_, v + (size_t)n0b * T_, T_, kmax, wm, wn,
                 acc);

  const int nl = lane & 15, mb = (lane >> 4) * 8;
#pragma unroll
  for (int i = 0; i < 2; ++i)
#pragma unroll
    for (int j = 0; j < 4; ++j) {
      const int n = n0 + 16 * j + nl, mrow = m0 + 16 * i + mb;
#pragma unroll
      for (int r = 0; r < 8; ++r)
        o[(size_t)(mrow + r) * H_ + n] = acc[i][j][r];
    }
}

// ---------------------------------------------------------------------------
extern "C" void kernel_launch(void* const* d_in, const int* in_sizes, int n_in,
                              void* d_out, int out_size, void* d_ws,
                              size_t ws_size, hipStream_t stream) {
  const float* x  = (const float*)d_in[0];
  const float* Wq = (const float*)d_in[1];
  const float* bq = (const float*)d_in[2];
  const float* Wk = (const float*)d_in[3];
  const float* bk = (const float*)d_in[4];
  const float* Wv = (const float*)d_in[5];
  const float* bv = (const float*)d_in[6];

  char* ws = (char*)d_ws;
  const size_t SX = (size_t)B_ * T_ * C_ * 2;   // xb  bf16   (67 MB)
  const size_t SW = (size_t)H_ * C_ * 2;        // W^T bf16   (8.4 MB each)
  const size_t SQ = (size_t)B_ * T_ * H_ * 2;   // q/k/vt bf16 (67 MB each)
  const size_t SS = (size_t)B_ * T_ * T_ * 4;   // S fp32     (134 MB)
  bhalf* xb  = (bhalf*)(ws);
  bhalf* wqT = (bhalf*)(ws + SX);
  bhalf* wkT = (bhalf*)(ws + SX + SW);
  bhalf* wvT = (bhalf*)(ws + SX + 2 * SW);
  bhalf* qb  = (bhalf*)(ws + SX + 3 * SW);
  bhalf* kb  = (bhalf*)(ws + SX + 3 * SW + SQ);
  bhalf* vtb = (bhalf*)(ws + SX + 3 * SW + 2 * SQ);
  float* Sws = (float*)(ws + SX + 3 * SW + 3 * SQ);
  bhalf* Pws = (bhalf*)(ws + SX + 3 * SW + 3 * SQ + SS);

  // 0a: x -> bf16
  {
    const int n = B_ * T_ * C_;
    k_cvt_bf16<<<n / 4 / 256, 256, 0, stream>>>(x, xb, n);
  }
  // 0b: transpose+convert weights
  {
    dim3 tb(32, 8), tg(H_ / 32, C_ / 32);
    k_transpose_w<<<tg, tb, 0, stream>>>(Wq, wqT);
    k_transpose_w<<<tg, tb, 0, stream>>>(Wk, wkT);
    k_transpose_w<<<tg, tb, 0, stream>>>(Wv, wvT);
  }
  // 1: q/k/v projections (v stored transposed [B,H,T])
  {
    dim3 pg(H_ / 128, (B_ * T_) / 128);
    k_proj<<<pg, 256, 0, stream>>>(xb, wqT, bq, qb, nullptr, 0);
    k_proj<<<pg, 256, 0, stream>>>(xb, wkT, bk, kb, nullptr, 0);
    k_proj<<<pg, 256, 0, stream>>>(xb, wvT, bv, nullptr, vtb, 1);
  }
  // 2: scores (causal blocks only)
  {
    const float scale = 1.0f / sqrtf((float)H_);
    dim3 sg(T_ / 128, T_ / 128, B_);
    k_scores<<<sg, 256, 0, stream>>>(qb, kb, Sws, scale);
  }
  // 3: causal softmax -> P bf16
  k_softmax<<<dim3(T_, B_), 256, 0, stream>>>(Sws, Pws);
  // 4: O = P @ V
  k_out<<<dim3(H_ / 128, T_ / 128, B_), 256, 0, stream>>>(Pws, vtb,
                                                          (float*)d_out);
}
